// Attention_5909875000179
// MI455X (gfx1250) — matrix operand order repaired and checked by reading, not compiled, not run on hardware
//
#include <hip/hip_runtime.h>
#include <math.h>

#define S_LEN 2048
#define B_SZ  32
#define H_SZ  1024
#define E_SZ  2048            // 2H, K dimension of the big GEMM

typedef __bf16 bf16_t;
typedef bf16_t v16bf __attribute__((ext_vector_type(16)));
typedef bf16_t v8bf  __attribute__((ext_vector_type(8)));
typedef bf16_t v2bf  __attribute__((ext_vector_type(2)));
typedef float  v8f   __attribute__((ext_vector_type(8)));

// pack two f32 into bf16x2 using the hardware convert
__device__ __forceinline__ unsigned pack2_bf16(float a, float b) {
  v2bf p;
  p.x = (bf16_t)a;
  p.y = (bf16_t)b;
  return __builtin_bit_cast(unsigned, p);
}

// fast tanh: (e^2x - 1) * rcp(e^2x + 1); v_exp_f32 + v_rcp_f32 (TRANS, co-executes)
__device__ __forceinline__ float fast_tanh(float x) {
  float cx = fminf(fmaxf(x, -12.0f), 12.0f);
  float e2 = __expf(2.0f * cx);
  return (e2 - 1.0f) * __builtin_amdgcn_rcpf(e2 + 1.0f);
}

__device__ __forceinline__ v16bf cat16(v8bf lo, v8bf hi) {
  return __builtin_shufflevector(lo, hi,
      0, 1, 2, 3, 4, 5, 6, 7, 8, 9, 10, 11, 12, 13, 14, 15);
}

// Scheduling pin for one K-step: interleave 34 DS reads (2 A + 32 B) with 16
// WMMAs so each WMMA overlaps the ds_loads of a future B tile.
// 0x100 = DS read group, 0x008 = MFMA/WMMA group.
__device__ __forceinline__ void sched_pipeline_ks() {
#if __has_builtin(__builtin_amdgcn_sched_group_barrier)
  __builtin_amdgcn_sched_group_barrier(0x100, 6, 0);   // A frag + B tiles 0,1
#pragma unroll
  for (int i = 0; i < 14; ++i) {
    __builtin_amdgcn_sched_group_barrier(0x008, 1, 0); // WMMA tile i
    __builtin_amdgcn_sched_group_barrier(0x100, 2, 0); // loads for tile i+2
  }
  __builtin_amdgcn_sched_group_barrier(0x008, 2, 0);   // last two WMMAs
#endif
}

// ---------------- Kernel 1: h_proj = hidden @ Wh + b  (32 x 1024) ----------------
__global__ void hproj_kernel(const float* __restrict__ hidden,
                             const float* __restrict__ W,
                             const float* __restrict__ bias,
                             float* __restrict__ hproj) {
  int t = blockIdx.x * blockDim.x + threadIdx.x;   // 0..32767
  int h = t & (H_SZ - 1);
  int b = t >> 10;
  const float* hrow = hidden + b * H_SZ;
  float sum = bias[h];
#pragma unroll 8
  for (int k = 0; k < H_SZ; ++k)
    sum = fmaf(hrow[k], W[k * H_SZ + h], sum);     // Wh = W[:H], column h coalesced
  hproj[t] = sum;
}

// ---------------- Kernel 2: fused  scores[b][s] = v . tanh(enc@We + h_proj) -------
#define KC      64                       // K chunk (2 WMMA K-steps)
#define LSTRIDE 72                       // padded LDS stride in halves (144B, bank-friendly)
#define LDS_A_OFF 0
#define LDS_A_SZ  (B_SZ * LSTRIDE * 2)               // 4608 B
#define LDS_B_OFF (LDS_A_OFF + LDS_A_SZ)
#define LDS_B_SZ  (H_SZ * LSTRIDE * 2)               // 147456 B
#define LDS_P_OFF (LDS_B_OFF + LDS_B_SZ)
#define LDS_TOTAL (LDS_P_OFF + B_SZ * 4)             // 152192 B  -> 2 WGs / WGP

// LDS limits occupancy to 2 WGs/WGP (4 waves/SIMD); tell the compiler VGPRs
// are not the constraint so the scheduler stops pressure-serializing the
// ds_load -> wmma pipeline.
__global__ __launch_bounds__(256, 2)
void fused_energy_kernel(const float* __restrict__ enc,   // (S, B, 2H) row-major
                         const float* __restrict__ W,     // (3H, H)
                         const float* __restrict__ vvec,  // (H)
                         const float* __restrict__ hproj, // (B, H)
                         float* __restrict__ scores) {    // (B, S) pre-softmax
  extern __shared__ char smem[];
  unsigned short* As      = (unsigned short*)(smem + LDS_A_OFF);
  unsigned short* Bs      = (unsigned short*)(smem + LDS_B_OFF);
  float*          partial = (float*)(smem + LDS_P_OFF);

  const int tid    = threadIdx.x;
  const int lane   = tid & 31;
  const int wave   = tid >> 5;
  const int wm     = wave >> 2;          // 0..1 : which 16-row half
  const int wn     = wave & 3;           // 0..3 : which 256-col quarter
  const int m_base = wm * 16;
  const int ln     = lane & 15;
  const int hi     = lane >> 4;
  const int s      = blockIdx.x;         // one s per block; 32 rows = all b

  if (tid < B_SZ) partial[tid] = 0.0f;

  const float* enc_blk = enc + (size_t)s * B_SZ * E_SZ;  // 32 rows x 2048, contiguous
  const float* We      = W + (size_t)H_SZ * H_SZ;        // W[H:], (2048 x 1024)

  v8f acc[16] = {};                       // 16 column-tiles of 16x16 f32 per wave

  for (int kc = 0; kc < E_SZ; kc += KC) {
    // ---- stage A: 32 x 64 f32 -> bf16 LDS, row-major, padded stride
    for (int u = tid; u < (B_SZ * KC / 4); u += 256) {   // 512 units, 2 per thread
      int row = u >> 4;
      int kg  = (u & 15) << 2;
      const float4 f = *(const float4*)(enc_blk + row * E_SZ + kc + kg);
      uint2 p;
      p.x = pack2_bf16(f.x, f.y);
      p.y = pack2_bf16(f.z, f.w);
      *(uint2*)(As + row * LSTRIDE + kg) = p;
    }
    // ---- stage B: 64 x 1024 f32 -> bf16 LDS, [n][k] layout (K contiguous per column)
    for (int u = tid; u < (H_SZ * KC / 4); u += 256) {   // 16384 units
      int n  = u & (H_SZ - 1);                           // consecutive lanes -> coalesced
      int kg = (u >> 10) << 2;
      const float* wp = We + (size_t)(kc + kg) * H_SZ + n;
      uint2 p;
      p.x = pack2_bf16(wp[0], wp[H_SZ]);
      p.y = pack2_bf16(wp[2 * H_SZ], wp[3 * H_SZ]);
      *(uint2*)(Bs + n * LSTRIDE + kg) = p;
    }
    // warm next chunk (global_prefetch_b8)
    if (kc + KC < E_SZ) {
      __builtin_prefetch(enc_blk + (tid >> 3) * E_SZ + kc + KC + ((tid & 7) << 3), 0, 1);
      __builtin_prefetch(We + (size_t)(kc + KC + (tid >> 6)) * H_SZ + ((tid & 63) << 4), 0, 1);
    }
    __syncthreads();

#pragma unroll
    for (int ks = 0; ks < KC; ks += 32) {
      // A fragment in this code's contiguous K order (not the ISA table order):
      // halves 0-15 : K = ks + 16*hi + 0..15 (contiguous, matches B)
      const unsigned short* ap = As + (m_base + ln) * LSTRIDE + ks + 16 * hi;
      v8bf alo = *(const v8bf*)ap;
      v8bf ahi = *(const v8bf*)(ap + 8);
      v16bf afrag = cat16(alo, ahi);

      // B fragments: lane column = ln, contiguous 16-K run at ks + 16*hi.
      // Ping-pong double buffering; sched_group_barrier pins the interleave.
      const unsigned short* bp = Bs + (wn * 256 + ln) * LSTRIDE + ks + 16 * hi;
      v8bf blo0 = *(const v8bf*)bp;
      v8bf bhi0 = *(const v8bf*)(bp + 8);
      v8bf blo1, bhi1;
#pragma unroll
      for (int t = 0; t < 16; t += 2) {
        // issue loads for tile t+1 (buffer 1) before consuming buffer 0
        const unsigned short* bp1 = bp + 16 * LSTRIDE;
        blo1 = *(const v8bf*)bp1;
        bhi1 = *(const v8bf*)(bp1 + 8);
        acc[t] = __builtin_amdgcn_wmma_f32_16x16x32_bf16(
            false, afrag, false, cat16(blo0, bhi0), (short)0, acc[t], false, false);
        bp += 32 * LSTRIDE;
        if (t + 2 < 16) {
          // issue loads for tile t+2 (buffer 0) before consuming buffer 1
          blo0 = *(const v8bf*)bp;
          bhi0 = *(const v8bf*)(bp + 8);
        }
        acc[t + 1] = __builtin_amdgcn_wmma_f32_16x16x32_bf16(
            false, afrag, false, cat16(blo1, bhi1), (short)0, acc[t + 1], false, false);
      }
      sched_pipeline_ks();
    }
    __syncthreads();
  }

  // ---- epilogue: e = acc + h_proj[b][n]; score_row += v[n]*tanh(e)
  float rowsum[8];
#pragma unroll
  for (int j = 0; j < 8; ++j) rowsum[j] = 0.0f;

#pragma unroll
  for (int t = 0; t < 16; ++t) {
    int n = wn * 256 + t * 16 + ln;                 // this lane's column (C/D layout)
    float vn = vvec[n];
    const float* hp = hproj + n;
#pragma unroll
    for (int j = 0; j < 8; ++j) {
      int m = m_base + j + 8 * hi;                  // row = b index (C/D VGPR layout)
      float e = acc[t][j] + hp[m * H_SZ];
      rowsum[j] += vn * fast_tanh(e);
    }
  }
#pragma unroll
  for (int j = 0; j < 8; ++j) {
    int m = m_base + j + 8 * hi;
    atomicAdd(&partial[m], rowsum[j]);              // ds_add_f32
  }
  __syncthreads();
  if (tid < B_SZ) scores[(size_t)tid * S_LEN + s] = partial[tid];
}

// ---------------- Kernel 3: in-place softmax over S for each b --------------------
__global__ void softmax_kernel(float* __restrict__ scores) {
  __shared__ float red[256];
  const int b = blockIdx.x;
  const int tid = threadIdx.x;
  float* row = scores + (size_t)b * S_LEN;
  float vals[8];
  float m = -INFINITY;
#pragma unroll
  for (int i = 0; i < 8; ++i) { vals[i] = row[tid + i * 256]; m = fmaxf(m, vals[i]); }
  red[tid] = m; __syncthreads();
  for (int off = 128; off > 0; off >>= 1) {
    if (tid < off) red[tid] = fmaxf(red[tid], red[tid + off]);
    __syncthreads();
  }
  m = red[0]; __syncthreads();
  float sum = 0.0f;
#pragma unroll
  for (int i = 0; i < 8; ++i) { vals[i] = __expf(vals[i] - m); sum += vals[i]; }
  red[tid] = sum; __syncthreads();
  for (int off = 128; off > 0; off >>= 1) {
    if (tid < off) red[tid] += red[tid + off];
    __syncthreads();
  }
  float inv = 1.0f / red[0];
#pragma unroll
  for (int i = 0; i < 8; ++i) row[tid + i * 256] = vals[i] * inv;
}

// ---------------------------------------------------------------------------------
extern "C" void kernel_launch(void* const* d_in, const int* in_sizes, int n_in,
                              void* d_out, int out_size, void* d_ws, size_t ws_size,
                              hipStream_t stream) {
  (void)in_sizes; (void)n_in; (void)out_size; (void)ws_size;
  const float* hidden = (const float*)d_in[0];   // (B, H)
  const float* enc    = (const float*)d_in[1];   // (S, B, 2H)
  const float* W      = (const float*)d_in[2];   // (3H, H)
  const float* bias   = (const float*)d_in[3];   // (H)
  const float* vvec   = (const float*)d_in[4];   // (H)
  float* out   = (float*)d_out;                  // (B, S)
  float* hproj = (float*)d_ws;                   // B*H f32 = 128 KB scratch

  hproj_kernel<<<(B_SZ * H_SZ) / 256, 256, 0, stream>>>(hidden, W, bias, hproj);
  fused_energy_kernel<<<S_LEN, 256, LDS_TOTAL, stream>>>(enc, W, vvec, hproj, out);
  softmax_kernel<<<B_SZ, 256, 0, stream>>>(out);
}
